// MultiHeadAttention_34144990003501
// MI455X (gfx1250) — compile-verified, matrix operand order checked by reading, not run on hardware
//
#include <hip/hip_runtime.h>
#include <hip/hip_bf16.h>
#include <math.h>

// Problem constants (match reference)
#define B_   8
#define S_   2048
#define H_   256
#define NH_  4
#define HD_  64
#define M_   (B_ * S_)          // 16384 rows

typedef __attribute__((ext_vector_type(16))) __bf16       v16bf;
typedef __attribute__((ext_vector_type(8)))  float        v8f;
typedef __attribute__((ext_vector_type(4)))  unsigned int u32x4;

struct U32x8 { u32x4 lo, hi; };

// float -> bf16 (round to nearest even, NaN-agnostic — inputs are sane)
static __device__ __forceinline__ unsigned short f2bf(float f) {
  unsigned u = __builtin_bit_cast(unsigned, f);
  u += 0x7FFFu + ((u >> 16) & 1u);
  return (unsigned short)(u >> 16);
}

// Build a WMMA 16-bf16 operand from two aligned 16-byte chunks.
// Per ISA 7.12.2 (wave32): per lane, elements 0..7 <-> K = half*8 + e,
// elements 8..15 <-> K = 16 + half*8 + e.  Storing tiles row-major [m][k]
// (A) or [n][k] (B-transposed) makes both chunks contiguous b128 loads.
static __device__ __forceinline__ v16bf ld16bf(const unsigned short* p0,
                                               const unsigned short* p1) {
  U32x8 u;
  u.lo = *reinterpret_cast<const u32x4*>(p0);
  u.hi = *reinterpret_cast<const u32x4*>(p1);
  return __builtin_bit_cast(v16bf, u);
}

static __device__ __forceinline__ v8f wmma_bf16(v16bf a, v16bf b, v8f c) {
  return __builtin_amdgcn_wmma_f32_16x16x32_bf16(
      /*neg_a=*/false, a, /*neg_b=*/false, b,
      /*c_mod=*/(short)0, c, /*reuse_a=*/false, /*reuse_b=*/false);
}

// --- CDNA5 async memory->LDS copy (ASYNCcnt path) --------------------------
// Per lane: LDS[ldsOff] <- MEM[base + gOff] (16 bytes).  GVS addressing:
// 64-bit SGPR base + 32-bit per-lane VGPR offset.  LDS offset is the low
// 32 bits of the generic shared-space address (aperture: addr[31:0]).
static __device__ __forceinline__ unsigned lds_off(const void* p) {
  return (unsigned)(size_t)p;
}
static __device__ __forceinline__ void async_lds_b128(unsigned ldsOff,
                                                      unsigned gOff,
                                                      const void* base) {
  asm volatile("global_load_async_to_lds_b128 %0, %1, %2"
               :
               : "v"(ldsOff), "v"(gOff),
                 "s"((unsigned long long)(size_t)base)
               : "memory");
}

// ---------------------------------------------------------------------------
// Kernel 1: transpose + convert the 4 weight matrices to bf16 [n][k]
// ---------------------------------------------------------------------------
__global__ __launch_bounds__(256) void prep_weights_kernel(
    const float* __restrict__ Wq, const float* __restrict__ Wk,
    const float* __restrict__ Wv, const float* __restrict__ Wo,
    unsigned short* __restrict__ wT) {
  const int mat = blockIdx.y;
  const float* src = (mat == 0) ? Wq : (mat == 1) ? Wk : (mat == 2) ? Wv : Wo;
  unsigned short* dst = wT + (size_t)mat * H_ * H_;
  const int e = blockIdx.x * 256 + threadIdx.x;   // e = k*256 + n
  const int k = e >> 8, n = e & 255;
  dst[n * H_ + k] = f2bf(src[e]);
}

// ---------------------------------------------------------------------------
// Kernel 2: QKV projection.  16-row m-tile per block, 16 waves = 16 n-tiles.
// Writes Q,K bf16 [b][h][s][d]; V transposed bf16 [b][h][d][s].
// ---------------------------------------------------------------------------
__global__ __launch_bounds__(512) void qkv_kernel(
    const float* __restrict__ x,
    const float* __restrict__ bq, const float* __restrict__ bk,
    const float* __restrict__ bv,
    const unsigned short* __restrict__ wqT,
    const unsigned short* __restrict__ wkT,
    const unsigned short* __restrict__ wvT,
    unsigned short* __restrict__ Q, unsigned short* __restrict__ K,
    unsigned short* __restrict__ Vt) {
  __shared__ __align__(16) unsigned short xA[16][264];

  const int tid   = threadIdx.x;
  const int mbase = blockIdx.x * 16;

  { // stage 16x256 fp32 -> bf16 LDS
    const int row = tid >> 5;
    const int col = (tid & 31) * 8;
    const float* src = x + (size_t)(mbase + row) * H_ + col;
#pragma unroll
    for (int i = 0; i < 8; ++i) xA[row][col + i] = f2bf(src[i]);
  }
  __syncthreads();

  const int wave = tid >> 5, lane = tid & 31;
  const int half = lane >> 4, ln = lane & 15;
  const int j = wave * 16 + ln;                // global output column

  v8f cq, ck, cv;
  const float biasq = bq[j], biask = bk[j], biasv = bv[j];
#pragma unroll
  for (int r = 0; r < 8; ++r) { cq[r] = biasq; ck[r] = biask; cv[r] = biasv; }

#pragma unroll
  for (int kk = 0; kk < H_; kk += 32) {
    v16bf a = ld16bf(&xA[ln][kk + half * 8], &xA[ln][kk + 16 + half * 8]);
    const unsigned short* pq = wqT + (size_t)j * H_ + kk + half * 8;
    const unsigned short* pk = wkT + (size_t)j * H_ + kk + half * 8;
    const unsigned short* pv = wvT + (size_t)j * H_ + kk + half * 8;
    v16bf bqv = ld16bf(pq, pq + 16);
    v16bf bkv = ld16bf(pk, pk + 16);
    v16bf bvv = ld16bf(pv, pv + 16);
    cq = wmma_bf16(a, bqv, cq);
    ck = wmma_bf16(a, bkv, ck);
    cv = wmma_bf16(a, bvv, cv);
  }

  const int h = j >> 6, d = j & 63;
#pragma unroll
  for (int r = 0; r < 8; ++r) {
    const int gr = mbase + r + 8 * half;          // global row in [M_]
    const int b = gr >> 11, s = gr & (S_ - 1);
    const size_t qoff = (((size_t)b * NH_ + h) * S_ + s) * HD_ + d;
    Q[qoff] = f2bf(cq[r]);
    K[qoff] = f2bf(ck[r]);
    Vt[(((size_t)b * NH_ + h) * HD_ + d) * S_ + s] = f2bf(cv[r]);
  }
}

// ---------------------------------------------------------------------------
// Kernel 3: flash attention.  Block = 4 waves, each wave owns a 16-query
// tile.  K/V streamed through double-buffered LDS chunks via CDNA5
// GLOBAL_LOAD_ASYNC_TO_LDS_B128 (ASYNCcnt), prefetching chunk i+1 while
// computing chunk i.
// ---------------------------------------------------------------------------
__global__ __launch_bounds__(128) void attn_kernel(
    const float* __restrict__ mask,
    const unsigned short* __restrict__ Q,
    const unsigned short* __restrict__ K,
    const unsigned short* __restrict__ Vt,
    unsigned short* __restrict__ ctx) {
  __shared__ __align__(16) unsigned short Qs[4][16][72];   // per-wave Q [m][d]
  __shared__ __align__(16) unsigned short Ks[2][32][72];   // key chunk [key][d]
  __shared__ __align__(16) unsigned short Vs[2][64][40];   // V^T chunk [d][key]
  __shared__ __align__(16) unsigned short Ps[4][16][40];   // per-wave P [m][key]

  const int tid = threadIdx.x, wave = tid >> 5, lane = tid & 31;
  const int half = lane >> 4, ln = lane & 15;
  const int b = blockIdx.z, h = blockIdx.y;
  const int qb = blockIdx.x * 64 + wave * 16;

  const size_t headQK = ((size_t)b * NH_ + h) * S_ * HD_;
  const size_t headV  = ((size_t)b * NH_ + h) * HD_ * S_;
  const unsigned short* qbase = Q + headQK + (size_t)qb * HD_;
  const unsigned short* kbase = K + headQK;
  const unsigned short* vbase = Vt + headV;

  // async-stage this wave's Q tile (4 wave-ops, retired by the first wait)
#pragma unroll
  for (int u = 0; u < 4; ++u) {
    const int idx = lane + u * 32;                 // 0..127 of 16x8 units
    const int row = idx >> 3, ch = (idx & 7) * 8;
    async_lds_b128(lds_off(&Qs[wave][row][ch]),
                   (unsigned)((row * HD_ + ch) * 2), qbase);
  }

  // issue one 32-key K/V chunk into buffer bf (4 wave-ops, in-order)
  auto issue_chunk = [&](int k0, int bf) {
#pragma unroll
    for (int u = 0; u < 2; ++u) {                  // K: 32x64 bf16
      const int idx = tid + u * 128;
      const int row = idx >> 3, ch = (idx & 7) * 8;
      async_lds_b128(lds_off(&Ks[bf][row][ch]),
                     (unsigned)(((k0 + row) * HD_ + ch) * 2), kbase);
    }
#pragma unroll
    for (int u = 0; u < 2; ++u) {                  // V^T: 64x32 bf16
      const int idx = tid + u * 128;
      const int row = idx >> 2, ch = (idx & 3) * 8;
      async_lds_b128(lds_off(&Vs[bf][row][ch]),
                     (unsigned)((row * S_ + k0 + ch) * 2), vbase);
    }
  };

  v8f o0, o1, o2, o3;
  float mrow[8], lrow[8];
#pragma unroll
  for (int r = 0; r < 8; ++r) {
    o0[r] = 0.f; o1[r] = 0.f; o2[r] = 0.f; o3[r] = 0.f;
    mrow[r] = -1e30f; lrow[r] = 0.f;
  }

  issue_chunk(0, 0);

  for (int k0 = 0; k0 < S_; k0 += 32) {
    const int bf = (k0 >> 5) & 1;
    if (k0 + 32 < S_) {
      issue_chunk(k0 + 32, bf ^ 1);
      // 4 ops of chunk i+1 in flight; retire chunk i (and Q on iter 0)
      asm volatile("s_wait_asynccnt 4" ::: "memory");
    } else {
      asm volatile("s_wait_asynccnt 0" ::: "memory");
    }
    __syncthreads();   // chunk bf visible to all waves

    // scores = Q . K^T over d=64 (two K=32 WMMA steps, two 16-key subtiles)
    v8f s0, s1;
#pragma unroll
    for (int r = 0; r < 8; ++r) { s0[r] = 0.f; s1[r] = 0.f; }
#pragma unroll
    for (int kk = 0; kk < 64; kk += 32) {
      v16bf a  = ld16bf(&Qs[wave][ln][kk + half * 8],
                        &Qs[wave][ln][kk + 16 + half * 8]);
      v16bf b0 = ld16bf(&Ks[bf][ln][kk + half * 8],
                        &Ks[bf][ln][kk + 16 + half * 8]);
      v16bf b1 = ld16bf(&Ks[bf][16 + ln][kk + half * 8],
                        &Ks[bf][16 + ln][kk + 16 + half * 8]);
      s0 = wmma_bf16(a, b0, s0);
      s1 = wmma_bf16(a, b1, s1);
    }

    // scale + attention mask
#pragma unroll
    for (int r = 0; r < 8; ++r) {
      const size_t moff = ((size_t)b * S_ + (qb + r + 8 * half)) * S_ + k0;
      s0[r] = s0[r] * 0.125f + mask[moff + ln];
      s1[r] = s1[r] * 0.125f + mask[moff + 16 + ln];
    }

    // online softmax (row stats reduced across the 16-lane column group)
#pragma unroll
    for (int r = 0; r < 8; ++r) {
      float v = fmaxf(s0[r], s1[r]);
      v = fmaxf(v, __shfl_xor(v, 8, 32));
      v = fmaxf(v, __shfl_xor(v, 4, 32));
      v = fmaxf(v, __shfl_xor(v, 2, 32));
      v = fmaxf(v, __shfl_xor(v, 1, 32));
      const float mn   = fmaxf(mrow[r], v);
      const float corr = __expf(mrow[r] - mn);
      const float p0   = __expf(s0[r] - mn);
      const float p1   = __expf(s1[r] - mn);
      Ps[wave][r + 8 * half][ln]      = f2bf(p0);
      Ps[wave][r + 8 * half][16 + ln] = f2bf(p1);
      float ss = p0 + p1;
      ss += __shfl_xor(ss, 8, 32);
      ss += __shfl_xor(ss, 4, 32);
      ss += __shfl_xor(ss, 2, 32);
      ss += __shfl_xor(ss, 1, 32);
      lrow[r] = lrow[r] * corr + ss;
      mrow[r] = mn;
      o0[r] *= corr; o1[r] *= corr; o2[r] *= corr; o3[r] *= corr;
    }

    // O += P . V   (K=32 over this key chunk, 4 output d-tiles)
    v16bf pa  = ld16bf(&Ps[wave][ln][half * 8], &Ps[wave][ln][16 + half * 8]);
    v16bf vb0 = ld16bf(&Vs[bf][ln][half * 8],      &Vs[bf][ln][16 + half * 8]);
    v16bf vb1 = ld16bf(&Vs[bf][16 + ln][half * 8], &Vs[bf][16 + ln][16 + half * 8]);
    v16bf vb2 = ld16bf(&Vs[bf][32 + ln][half * 8], &Vs[bf][32 + ln][16 + half * 8]);
    v16bf vb3 = ld16bf(&Vs[bf][48 + ln][half * 8], &Vs[bf][48 + ln][16 + half * 8]);
    o0 = wmma_bf16(pa, vb0, o0);
    o1 = wmma_bf16(pa, vb1, o1);
    o2 = wmma_bf16(pa, vb2, o2);
    o3 = wmma_bf16(pa, vb3, o3);

    __syncthreads();   // all waves done with buffer bf before it refills
  }

  // normalize and emit context, merged heads bf16 [b*S+s][H]
#pragma unroll
  for (int r = 0; r < 8; ++r) {
    const float inv = 1.0f / lrow[r];
    const int srow = qb + r + 8 * half;
    const size_t crow = ((size_t)b * S_ + srow) * H_ + (size_t)h * HD_;
    ctx[crow + ln]      = f2bf(o0[r] * inv);
    ctx[crow + 16 + ln] = f2bf(o1[r] * inv);
    ctx[crow + 32 + ln] = f2bf(o2[r] * inv);
    ctx[crow + 48 + ln] = f2bf(o3[r] * inv);
  }
}

// ---------------------------------------------------------------------------
// Kernel 4: output projection + bias + residual + LayerNorm (fused)
// ---------------------------------------------------------------------------
__global__ __launch_bounds__(512) void proj_ln_kernel(
    const unsigned short* __restrict__ ctx,
    const unsigned short* __restrict__ woT,
    const float* __restrict__ bo, const float* __restrict__ x,
    const float* __restrict__ gamma, const float* __restrict__ beta,
    float* __restrict__ out) {
  __shared__ __align__(16) unsigned short cA[16][264];
  __shared__ __align__(16) float          ot[16][264];

  const int tid = threadIdx.x;
  const int mbase = blockIdx.x * 16;
  { // stage ctx tile (already bf16)
    const int row = tid >> 5, ch = (tid & 31) * 8;
    *reinterpret_cast<u32x4*>(&cA[row][ch]) =
        *reinterpret_cast<const u32x4*>(ctx + (size_t)(mbase + row) * H_ + ch);
  }
  __syncthreads();

  const int wave = tid >> 5, lane = tid & 31;
  const int half = lane >> 4, ln = lane & 15;
  const int j = wave * 16 + ln;

  v8f c;
  const float bb = bo[j];
#pragma unroll
  for (int r = 0; r < 8; ++r) c[r] = bb;

#pragma unroll
  for (int kk = 0; kk < H_; kk += 32) {
    v16bf a = ld16bf(&cA[ln][kk + half * 8], &cA[ln][kk + 16 + half * 8]);
    const unsigned short* bp = woT + (size_t)j * H_ + kk + half * 8;
    v16bf bv = ld16bf(bp, bp + 16);
    c = wmma_bf16(a, bv, c);
  }

  // + residual, stage rows for LayerNorm
#pragma unroll
  for (int r = 0; r < 8; ++r) {
    const int gr = mbase + r + 8 * half;
    ot[r + 8 * half][j] = c[r] + x[(size_t)gr * H_ + j];
  }
  __syncthreads();

  // LayerNorm: wave w normalizes row w (256 cols, 8 per lane)
  {
    const int row = wave, c0 = lane * 8;
    float vals[8], s = 0.f, sq = 0.f;
#pragma unroll
    for (int i = 0; i < 8; ++i) {
      vals[i] = ot[row][c0 + i];
      s += vals[i];
      sq += vals[i] * vals[i];
    }
    s  += __shfl_xor(s, 16, 32);  sq += __shfl_xor(sq, 16, 32);
    s  += __shfl_xor(s, 8, 32);   sq += __shfl_xor(sq, 8, 32);
    s  += __shfl_xor(s, 4, 32);   sq += __shfl_xor(sq, 4, 32);
    s  += __shfl_xor(s, 2, 32);   sq += __shfl_xor(sq, 2, 32);
    s  += __shfl_xor(s, 1, 32);   sq += __shfl_xor(sq, 1, 32);
    const float mu  = s * (1.0f / 256.0f);
    const float var = sq * (1.0f / 256.0f) - mu * mu;
    const float inv = rsqrtf(var + 1e-12f);
    const size_t gr = (size_t)(mbase + row) * H_;
#pragma unroll
    for (int i = 0; i < 8; ++i) {
      const int col = c0 + i;
      out[gr + col] = (vals[i] - mu) * inv * gamma[col] + beta[col];
    }
  }
}

// ---------------------------------------------------------------------------
extern "C" void kernel_launch(void* const* d_in, const int* in_sizes, int n_in,
                              void* d_out, int out_size, void* d_ws, size_t ws_size,
                              hipStream_t stream) {
  const float* x     = (const float*)d_in[0];
  const float* mask  = (const float*)d_in[1];
  const float* Wq    = (const float*)d_in[2];
  const float* bq    = (const float*)d_in[3];
  const float* Wk    = (const float*)d_in[4];
  const float* bk    = (const float*)d_in[5];
  const float* Wv    = (const float*)d_in[6];
  const float* bv    = (const float*)d_in[7];
  const float* Wo    = (const float*)d_in[8];
  const float* bo    = (const float*)d_in[9];
  const float* gamma = (const float*)d_in[10];
  const float* beta  = (const float*)d_in[11];
  float* out = (float*)d_out;

  // workspace layout (ushort elements)
  unsigned short* ws  = (unsigned short*)d_ws;
  unsigned short* wT  = ws;                         // 4 * 256*256
  unsigned short* Qb  = ws + 4 * H_ * H_;           // [B][NH][S][HD]
  unsigned short* Kb  = Qb + (size_t)B_ * NH_ * S_ * HD_;
  unsigned short* Vtb = Kb + (size_t)B_ * NH_ * S_ * HD_;
  unsigned short* ctx = Vtb + (size_t)B_ * NH_ * S_ * HD_;  // [M][H]

  prep_weights_kernel<<<dim3(H_ * H_ / 256, 4), 256, 0, stream>>>(
      Wq, Wk, Wv, Wo, wT);

  qkv_kernel<<<M_ / 16, 512, 0, stream>>>(
      x, bq, bk, bv, wT, wT + H_ * H_, wT + 2 * H_ * H_, Qb, Kb, Vtb);

  attn_kernel<<<dim3(S_ / 64, NH_, B_), 128, 0, stream>>>(
      mask, Qb, Kb, Vtb, ctx);

  proj_ln_kernel<<<M_ / 16, 512, 0, stream>>>(
      ctx, wT + 3 * H_ * H_, bo, x, gamma, beta, out);
}
